// Attention_61770219651786
// MI455X (gfx1250) — matrix operand order repaired and checked by reading, not compiled, not run on hardware
//
#include <hip/hip_runtime.h>
#include <hip/hip_bf16.h>
#include <math.h>

// ---------------- problem constants ----------------
#define B_  2
#define S_  2048
#define H_  2048
#define NH_ 16
#define KVH_ 4
#define HD_ 128
#define R_  16
#define QDIM (NH_*HD_)    // 2048
#define KVDIM (KVH_*HD_)  // 512
#define LORA_SCALE 2.0f
#define ATTN_SCALE 0.08838834764831843f  // 1/sqrt(128)
#define NEG_ -1000000000.0f

// GEMM register blocking: each wave computes (16*GM_) x (16*GN_) of C
#define GM_ 2
#define GN_ 4

#ifndef __has_builtin
#define __has_builtin(x) 0
#endif
#if __has_builtin(__builtin_amdgcn_tensor_load_to_lds) && __has_builtin(__builtin_amdgcn_s_wait_tensorcnt)
#define HAVE_TDM 1
#else
#define HAVE_TDM 0
#endif

typedef unsigned short u16;
typedef __attribute__((ext_vector_type(16))) __bf16 v16bf;
typedef __attribute__((ext_vector_type(8)))  __bf16 v8bf;
typedef __attribute__((ext_vector_type(8)))  float  v8f;
typedef __attribute__((ext_vector_type(4)))  unsigned int u32x4;
typedef __attribute__((ext_vector_type(8)))  int i32x8;
typedef __attribute__((ext_vector_type(4)))  int i32x4;

// float -> bf16 bits, round-to-nearest-even
__device__ __forceinline__ u16 f2bf(float f) {
  union { float f; unsigned u; } v; v.f = f;
  unsigned r = v.u + 0x7FFFu + ((v.u >> 16) & 1u);
  return (u16)(r >> 16);
}

// ---- WMMA fragment loaders (contiguous K order shared by A and B, wave32) ----
// A (16x32 bf16, MxK): lane&15 = row M. lanes 0-15 hold K=k0..k0+15,
// lanes 16-31 hold K=k0+16..k0+31 (16 consecutive K per lane).
__device__ __forceinline__ v16bf load_a_frag(const u16* base, int ld, int m, int k0, int lane) {
  const u16* p = base + (size_t)m * ld + k0 + ((lane < 16) ? 0 : 16);
  v8bf lo = *(const v8bf*)(p);
  v8bf hi = *(const v8bf*)(p + 8);
  return __builtin_shufflevector(lo, hi, 0,1,2,3,4,5,6,7,8,9,10,11,12,13,14,15);
}
// B (32x16 bf16, KxN): lane&15 = column N; lanes 0-15 hold K=k0..k0+15,
// lanes 16-31 hold K=k0+16..k0+31 (16 consecutive K per lane).
// B[k][n] is stored K-contiguous at rowbase[n*ld + k] (i.e. W row n, or K/V^T row).
__device__ __forceinline__ v16bf load_b_frag(const u16* rowbase, int ld, int n, int k0, int lane) {
  const u16* p = rowbase + (size_t)n * ld + k0 + ((lane < 16) ? 0 : 16);
  return *(const v16bf*)(p);
}

__device__ __forceinline__ v8f wmma_bf16(v16bf a, v16bf b, v8f c) {
  return __builtin_amdgcn_wmma_f32_16x16x32_bf16(false, a, false, b, (short)0, c, false, false);
}

#if HAVE_TDM
// Issue a TDM 2-D tile load: tile_y rows x tile_x bf16 elems, row stride
// stride_elems, packed densely into LDS at lds_off (row-major, tile_x*2 B/row).
// D# packing per CDNA5 ISA 8.3/8.4 (group0 128b, group1 256b; groups 2/3 zero).
// Toolchain uses the 6-arg builtin: (g0, g1, g2, g3, g4, cpol).
__device__ __forceinline__ void tdm_load_tile_bf16(unsigned lds_off, const void* gaddr,
                                                   unsigned tile_x, unsigned tile_y,
                                                   unsigned stride_elems) {
  unsigned long long ga = (unsigned long long)gaddr;
  u32x4 g0;
  g0[0] = 1u;                                          // count=1, user-mode
  g0[1] = lds_off;                                     // lds_addr (bytes)
  g0[2] = (unsigned)(ga & 0xffffffffu);                // global_addr[31:0]
  g0[3] = (unsigned)((ga >> 32) & 0x1ffffffu) | (2u << 30);  // addr[56:32] | type=2
  i32x8 g1;
  g1[0] = (int)(1u << 16);                             // wg_mask=0, data_size=1 (2B)
  g1[1] = (int)((tile_x & 0xffffu) << 16);             // tensor_dim0[15:0]
  g1[2] = (int)((tile_x >> 16) | ((tile_y & 0xffffu) << 16)); // dim0 hi | dim1 lo
  g1[3] = (int)((tile_y >> 16) | (tile_x << 16));      // dim1 hi | tile_dim0
  g1[4] = (int)(tile_y & 0xffffu);                     // tile_dim1 (tile_dim2=0)
  g1[5] = (int)stride_elems;                           // tensor_dim0_stride[31:0]
  g1[6] = 0;                                           // stride[47:32]=0 | dim1_stride lo
  g1[7] = 0;
  i32x4 gz4 = {0, 0, 0, 0};
  i32x8 gz8 = {0, 0, 0, 0, 0, 0, 0, 0};
  __builtin_amdgcn_tensor_load_to_lds(g0, g1, gz4, gz4, gz8, 0);
}
#endif

// ---------------- kernel 1: fuse LoRA into weights, convert to bf16 ----------------
// out[n][k] = bf16( W[n][k] + LORA_SCALE * sum_r A[k][r] * Bm[r][n] )
__global__ void fuse_weights_kernel(const float* __restrict__ W, const float* __restrict__ A,
                                    const float* __restrict__ Bm, u16* __restrict__ out,
                                    int N, int K) {
  int idx = blockIdx.x * blockDim.x + threadIdx.x;
  if (idx >= N * K) return;
  int n = idx / K, k = idx % K;
  float s = 0.f;
#pragma unroll
  for (int r = 0; r < R_; ++r) s += A[k * R_ + r] * Bm[r * N + n];
  out[idx] = f2bf(W[idx] + LORA_SCALE * s);
}

__global__ void convert_bf16_kernel(const float* __restrict__ in, u16* __restrict__ out, int n) {
  int idx = blockIdx.x * blockDim.x + threadIdx.x;
  if (idx < n) out[idx] = f2bf(in[idx]);
}

// ---------------- kernel 2: bf16 WMMA GEMM, C = X @ W^T + bias (f32 out) ----------------
// X: [M][K] bf16, W: [N][K] bf16. One wave computes a (16*GM_) x (16*GN_) macro-tile.
// With TDM: the 64x32 W-chunk per K-step is DMA'd into LDS (double-buffered,
// TENSORcnt-tracked, in-order), so the VMEM pipe only carries the A stream.
__global__ __launch_bounds__(32) void gemm_bf16_kernel(const u16* __restrict__ X,
    const u16* __restrict__ W, const float* __restrict__ bias,
    float* __restrict__ C, int M, int N, int K) {
#if HAVE_TDM
  __shared__ __align__(64) u16 wbuf[2][GN_ * 16 * 32];   // 2 x 4KB
#endif
  int lane = threadIdx.x;
  int m0 = blockIdx.x * (16 * GM_);
  int n0 = blockIdx.y * (16 * GN_);

  v8f acc[GM_][GN_];
#pragma unroll
  for (int i = 0; i < GM_; ++i)
#pragma unroll
    for (int j = 0; j < GN_; ++j)
      acc[i][j] = (v8f){0.f,0.f,0.f,0.f,0.f,0.f,0.f,0.f};

#if HAVE_TDM
  const u16* wtile = W + (size_t)n0 * K;   // 64 rows starting at n0, row stride K
  unsigned lds0 = (unsigned)(unsigned long long)(&wbuf[0][0]);
  unsigned lds1 = (unsigned)(unsigned long long)(&wbuf[1][0]);
  tdm_load_tile_bf16(lds0, wtile, 32, GN_ * 16, (unsigned)K);
  int cur = 0;
  for (int k0 = 0; k0 < K; k0 += 32, cur ^= 1) {
    if (k0 + 32 < K) {
      tdm_load_tile_bf16(cur ? lds0 : lds1, wtile + (k0 + 32), 32, GN_ * 16, (unsigned)K);
      __builtin_amdgcn_s_wait_tensorcnt(1);   // in-order TDM: current buffer landed
    } else {
      __builtin_amdgcn_s_wait_tensorcnt(0);
    }
    asm volatile("" ::: "memory");            // LDS now holds DMA'd data; no hoisting
    v16bf a[GM_], b[GN_];
#pragma unroll
    for (int i = 0; i < GM_; ++i) {
      a[i] = load_a_frag(X, K, m0 + i * 16 + (lane & 15), k0, lane);
      __builtin_prefetch(X + (size_t)(m0 + i * 16 + (lane & 15)) * K + k0 + 64, 0, 1);
    }
#pragma unroll
    for (int j = 0; j < GN_; ++j) {
      const u16* p = &wbuf[cur][(j * 16 + (lane & 15)) * 32 + ((lane < 16) ? 0 : 16)];
      b[j] = *(const v16bf*)p;
    }
#pragma unroll
    for (int i = 0; i < GM_; ++i)
#pragma unroll
      for (int j = 0; j < GN_; ++j)
        acc[i][j] = wmma_bf16(a[i], b[j], acc[i][j]);
    asm volatile("" ::: "memory");            // keep ds reads inside this iteration
  }
#else
  for (int k0 = 0; k0 < K; k0 += 32) {
    v16bf a[GM_], b[GN_];
#pragma unroll
    for (int i = 0; i < GM_; ++i)
      a[i] = load_a_frag(X, K, m0 + i * 16 + (lane & 15), k0, lane);
#pragma unroll
    for (int j = 0; j < GN_; ++j)
      b[j] = load_b_frag(W, K, n0 + j * 16 + (lane & 15), k0, lane);
#pragma unroll
    for (int i = 0; i < GM_; ++i)
#pragma unroll
      for (int j = 0; j < GN_; ++j)
        acc[i][j] = wmma_bf16(a[i], b[j], acc[i][j]);
  }
#endif

  int rsub = (lane < 16) ? 0 : 8;
#pragma unroll
  for (int j = 0; j < GN_; ++j) {
    int ncol = n0 + j * 16 + (lane & 15);
    float bv = bias ? bias[ncol] : 0.f;
#pragma unroll
    for (int i = 0; i < GM_; ++i) {
      int rbase = m0 + i * 16 + rsub;
#pragma unroll
      for (int v = 0; v < 8; ++v)
        C[(size_t)(rbase + v) * N + ncol] = acc[i][j][v] + bv;
    }
  }
}

// ---------------- kernel 3: RoPE + layout transforms ----------------
// q_raw:[B*S][QDIM], k_raw/v_raw:[B*S][KVDIM] f32.
// qout:[B*NH][S][HD] bf16, kout:[B*KVH][S][HD] bf16, vt:[B*KVH][HD][S] bf16.
__global__ void rope_kernel(const float* __restrict__ qraw, const float* __restrict__ kraw,
                            const float* __restrict__ vraw, const float* __restrict__ cosb,
                            const float* __restrict__ sinb, u16* __restrict__ qout,
                            u16* __restrict__ kout, u16* __restrict__ vt) {
  int bs = blockIdx.x;
  int b = bs / S_, s = bs % S_;
  const float* crow = cosb + ((size_t)b * S_ + s) * HD_;
  const float* srow = sinb + ((size_t)b * S_ + s) * HD_;
  const float* qr = qraw + (size_t)bs * QDIM;
  const float* kr = kraw + (size_t)bs * KVDIM;
  const float* vr = vraw + (size_t)bs * KVDIM;
  for (int i = threadIdx.x; i < QDIM; i += blockDim.x) {
    int h = i / HD_, d = i % HD_;
    float rot = (d < HD_/2) ? -qr[h * HD_ + d + HD_/2] : qr[h * HD_ + d - HD_/2];
    float val = qr[i] * crow[d] + rot * srow[d];
    qout[(((size_t)b * NH_ + h) * S_ + s) * HD_ + d] = f2bf(val);
  }
  for (int i = threadIdx.x; i < KVDIM; i += blockDim.x) {
    int h = i / HD_, d = i % HD_;
    float rot = (d < HD_/2) ? -kr[h * HD_ + d + HD_/2] : kr[h * HD_ + d - HD_/2];
    float val = kr[i] * crow[d] + rot * srow[d];
    kout[(((size_t)b * KVH_ + h) * S_ + s) * HD_ + d] = f2bf(val);
  }
  for (int i = threadIdx.x; i < KVDIM; i += blockDim.x) {
    int h = i / HD_, d = i % HD_;
    vt[(((size_t)b * KVH_ + h) * HD_ + d) * S_ + s] = f2bf(vr[i]);
  }
}

// ---------------- kernel 4: causal flash attention (WMMA, online softmax) ----------------
// One wave handles 16 query rows x full HD=128 for one (b,h). 32 keys per step.
__global__ __launch_bounds__(32) void flash_attn_kernel(
    const u16* __restrict__ Q,  // [B*NH][S][HD]
    const u16* __restrict__ Kc, // [B*KVH][S][HD]
    const u16* __restrict__ Vt, // [B*KVH][HD][S]
    u16* __restrict__ O) {      // [B*S][NH*HD]
  __shared__ u16 p_lds[16 * 32];
  int lane = threadIdx.x;
  int bh = blockIdx.x;          // b*NH + h
  int qt = blockIdx.y;          // query tile (16 rows)
  int b = bh / NH_, h = bh % NH_;
  int kvh = h / (NH_ / KVH_);
  const u16* qbase = Q + (size_t)bh * S_ * HD_;
  const u16* kbase = Kc + ((size_t)(b * KVH_ + kvh)) * S_ * HD_;
  const u16* vbase = Vt + ((size_t)(b * KVH_ + kvh)) * HD_ * S_;

  int colb = lane & 15;
  int mrow = qt * 16 + colb;
  int row0 = qt * 16 + ((lane < 16) ? 0 : 8);

  // preload Q A-fragments over all 4 K-chunks of HD=128
  v16bf qf[4];
#pragma unroll
  for (int c = 0; c < 4; ++c) qf[c] = load_a_frag(qbase, HD_, mrow, c * 32, lane);

  float mI[8], lI[8];
  v8f acc[8];
#pragma unroll
  for (int v = 0; v < 8; ++v) { mI[v] = -1e30f; lI[v] = 0.f; }
#pragma unroll
  for (int t = 0; t < 8; ++t) acc[t] = (v8f){0.f,0.f,0.f,0.f,0.f,0.f,0.f,0.f};

  int nblocks = ((qt + 1) * 16 + 31) / 32;
  for (int j = 0; j < nblocks; ++j) {
    int kb = j * 32;
    // scores: two 16x16 tiles over this key block
    v8f s0 = (v8f){0.f,0.f,0.f,0.f,0.f,0.f,0.f,0.f};
    v8f s1 = (v8f){0.f,0.f,0.f,0.f,0.f,0.f,0.f,0.f};
#pragma unroll
    for (int c = 0; c < 4; ++c) {
      v16bf kf0 = load_b_frag(kbase, HD_, kb + colb, c * 32, lane);
      s0 = wmma_bf16(qf[c], kf0, s0);
    }
#pragma unroll
    for (int c = 0; c < 4; ++c) {
      v16bf kf1 = load_b_frag(kbase, HD_, kb + 16 + colb, c * 32, lane);
      s1 = wmma_bf16(qf[c], kf1, s1);
    }
    // mask + online softmax update
    float p0[8], p1[8], alpha[8];
#pragma unroll
    for (int v = 0; v < 8; ++v) {
      int row = row0 + v;
      float a0 = s0[v] * ATTN_SCALE + ((kb + colb)      > row ? NEG_ : 0.f);
      float a1 = s1[v] * ATTN_SCALE + ((kb + 16 + colb) > row ? NEG_ : 0.f);
      float rm = fmaxf(a0, a1);
#pragma unroll
      for (int off = 1; off < 16; off <<= 1) rm = fmaxf(rm, __shfl_xor(rm, off, 32));
      float mnew = fmaxf(mI[v], rm);
      alpha[v] = __expf(mI[v] - mnew);
      mI[v] = mnew;
      p0[v] = __expf(a0 - mnew);
      p1[v] = __expf(a1 - mnew);
      float rs = p0[v] + p1[v];
#pragma unroll
      for (int off = 1; off < 16; off <<= 1) rs += __shfl_xor(rs, off, 32);
      lI[v] = lI[v] * alpha[v] + rs;
    }
#pragma unroll
    for (int t = 0; t < 8; ++t)
#pragma unroll
      for (int v = 0; v < 8; ++v) acc[t][v] *= alpha[v];
    // transpose P (C layout -> A layout) through LDS
#pragma unroll
    for (int v = 0; v < 8; ++v) {
      int row = ((lane < 16) ? 0 : 8) + v;
      p_lds[row * 32 + colb]      = f2bf(p0[v]);
      p_lds[row * 32 + 16 + colb] = f2bf(p1[v]);
    }
    __syncthreads();
    v16bf pA;
    {
      const u16* p = &p_lds[colb * 32 + ((lane < 16) ? 0 : 16)];
      v8bf lo = *(const v8bf*)p;
      v8bf hi = *(const v8bf*)(p + 8);
      pA = __builtin_shufflevector(lo, hi, 0,1,2,3,4,5,6,7,8,9,10,11,12,13,14,15);
    }
    __syncthreads();
    // P @ V: 8 output column tiles of 16
#pragma unroll
    for (int t = 0; t < 8; ++t) {
      const u16* p = vbase + (size_t)(t * 16 + colb) * S_ + kb + ((lane < 16) ? 0 : 16);
      v16bf vf = *(const v16bf*)p;
      acc[t] = wmma_bf16(pA, vf, acc[t]);
    }
  }
  // epilogue: normalize and store bf16 to [B*S][NH*HD]
#pragma unroll
  for (int t = 0; t < 8; ++t) {
#pragma unroll
    for (int v = 0; v < 8; ++v) {
      int row = row0 + v;
      float o = acc[t][v] / lI[v];
      O[((size_t)b * S_ + row) * QDIM + h * HD_ + t * 16 + colb] = f2bf(o);
    }
  }
}

// ---------------- host launcher ----------------
static inline size_t align256(size_t x) { return (x + 255) & ~(size_t)255; }

extern "C" void kernel_launch(void* const* d_in, const int* in_sizes, int n_in,
                              void* d_out, int out_size, void* d_ws, size_t ws_size,
                              hipStream_t stream) {
  const float* x    = (const float*)d_in[0];
  const float* cosb = (const float*)d_in[1];
  const float* sinb = (const float*)d_in[2];
  // d_in[3] = attention_mask (recomputed causally on device)
  const float* q_w = (const float*)d_in[4];
  const float* k_w = (const float*)d_in[5];
  const float* v_w = (const float*)d_in[6];
  const float* q_b = (const float*)d_in[7];
  const float* k_b = (const float*)d_in[8];
  const float* v_b = (const float*)d_in[9];
  const float* q_A = (const float*)d_in[10];
  const float* q_Bm = (const float*)d_in[11];
  const float* k_A = (const float*)d_in[12];
  const float* k_Bm = (const float*)d_in[13];
  const float* v_A = (const float*)d_in[14];
  const float* v_Bm = (const float*)d_in[15];
  const float* o_w = (const float*)d_in[16];
  float* out = (float*)d_out;

  const int M = B_ * S_;  // 4096
  char* ws = (char*)d_ws;
  size_t off = 0;
  u16* xb   = (u16*)(ws + off); off = align256(off + (size_t)M * H_ * 2);
  u16* qw   = (u16*)(ws + off); off = align256(off + (size_t)QDIM * H_ * 2);
  u16* kw   = (u16*)(ws + off); off = align256(off + (size_t)KVDIM * H_ * 2);
  u16* vw   = (u16*)(ws + off); off = align256(off + (size_t)KVDIM * H_ * 2);
  u16* ow   = (u16*)(ws + off); off = align256(off + (size_t)H_ * H_ * 2);
  float* qraw = (float*)(ws + off); off = align256(off + (size_t)M * QDIM * 4);
  float* kraw = (float*)(ws + off); off = align256(off + (size_t)M * KVDIM * 4);
  float* vraw = (float*)(ws + off); off = align256(off + (size_t)M * KVDIM * 4);
  u16* qrope = (u16*)(ws + off); off = align256(off + (size_t)B_ * NH_ * S_ * HD_ * 2);
  u16* krope = (u16*)(ws + off); off = align256(off + (size_t)B_ * KVH_ * S_ * HD_ * 2);
  u16* vt    = (u16*)(ws + off); off = align256(off + (size_t)B_ * KVH_ * HD_ * S_ * 2);
  u16* aout  = (u16*)(ws + off); off = align256(off + (size_t)M * QDIM * 2);

  // 1) prep: fuse LoRA into weights, convert x & o_w to bf16
  {
    int n = QDIM * H_;
    fuse_weights_kernel<<<(n + 255) / 256, 256, 0, stream>>>(q_w, q_A, q_Bm, qw, QDIM, H_);
    n = KVDIM * H_;
    fuse_weights_kernel<<<(n + 255) / 256, 256, 0, stream>>>(k_w, k_A, k_Bm, kw, KVDIM, H_);
    fuse_weights_kernel<<<(n + 255) / 256, 256, 0, stream>>>(v_w, v_A, v_Bm, vw, KVDIM, H_);
    n = M * H_;
    convert_bf16_kernel<<<(n + 255) / 256, 256, 0, stream>>>(x, xb, n);
    n = H_ * H_;
    convert_bf16_kernel<<<(n + 255) / 256, 256, 0, stream>>>(o_w, ow, n);
  }

  // 2) QKV projections (bf16 WMMA GEMM + TDM-staged weights, f32 out + bias)
  gemm_bf16_kernel<<<dim3(M / (16*GM_), QDIM  / (16*GN_)), 32, 0, stream>>>(xb, qw, q_b, qraw, M, QDIM, H_);
  gemm_bf16_kernel<<<dim3(M / (16*GM_), KVDIM / (16*GN_)), 32, 0, stream>>>(xb, kw, k_b, kraw, M, KVDIM, H_);
  gemm_bf16_kernel<<<dim3(M / (16*GM_), KVDIM / (16*GN_)), 32, 0, stream>>>(xb, vw, v_b, vraw, M, KVDIM, H_);

  // 3) RoPE + head-major / transposed-V layouts (bf16)
  rope_kernel<<<B_ * S_, 256, 0, stream>>>(qraw, kraw, vraw, cosb, sinb, qrope, krope, vt);

  // 4) causal flash attention
  flash_attn_kernel<<<dim3(B_ * NH_, S_ / 16), 32, 0, stream>>>(qrope, krope, vt, aout);

  // 5) output projection -> f32 d_out
  gemm_bf16_kernel<<<dim3(M / (16*GM_), H_ / (16*GN_)), 32, 0, stream>>>(aout, ow, nullptr, out, M, H_, H_);
}
